// MixtralMoE_60919816126694
// MI455X (gfx1250) — matrix operand order repaired and checked by reading, not compiled, not run on hardware
//
#include <hip/hip_runtime.h>
#include <hip/hip_bf16.h>
#include <cstdint>

// ---------------- problem constants ----------------
#define E_    8            // experts
#define HID   2048         // hidden
#define ITR   7168         // intermediate
#define TOK   8192         // B*S tokens
#define TOPK  2
// ---------------- tiling ----------------
#define MT    64           // slot (token) tile
#define NT    128          // output-column tile
#define KB    32           // WMMA K step (bf16)
#define CHS   2048         // slots per chunk (act buffer rows)
#define SLOT_MAX (TOK*TOPK + E_*MT)       // 16896 padded slots
#define NCHUNK   ((SLOT_MAX + CHS - 1)/CHS)

typedef __attribute__((ext_vector_type(16))) __bf16 v16bf;
typedef __attribute__((ext_vector_type(8)))  float  v8f;
typedef __attribute__((ext_vector_type(4)))  uint32_t u32x4;
typedef __attribute__((ext_vector_type(8)))  int      i32x8;
typedef __attribute__((ext_vector_type(4)))  int      i32x4;

union BFQ { v16bf v; uint4 q[2]; };

// Load 16 consecutive fp32 and convert to 16 bf16 (one B-fragment per lane).
__device__ __forceinline__ v16bf cvt16(const float* __restrict__ p) {
  float4 a = *(const float4*)(p);
  float4 b = *(const float4*)(p + 4);
  float4 c = *(const float4*)(p + 8);
  float4 d = *(const float4*)(p + 12);
  v16bf r;
  r[0]=(__bf16)a.x;  r[1]=(__bf16)a.y;  r[2]=(__bf16)a.z;  r[3]=(__bf16)a.w;
  r[4]=(__bf16)b.x;  r[5]=(__bf16)b.y;  r[6]=(__bf16)b.z;  r[7]=(__bf16)b.w;
  r[8]=(__bf16)c.x;  r[9]=(__bf16)c.y;  r[10]=(__bf16)c.z; r[11]=(__bf16)c.w;
  r[12]=(__bf16)d.x; r[13]=(__bf16)d.y; r[14]=(__bf16)d.z; r[15]=(__bf16)d.w;
  return r;
}

// Async global->LDS 16B copy (ASYNCcnt tracked, no VGPR round-trip).
__device__ __forceinline__ void async_b128(uint32_t ldsoff, uint64_t gaddr) {
  asm volatile("global_load_async_to_lds_b128 %0, %1, off"
               :: "v"(ldsoff), "v"(gaddr) : "memory");
}
__device__ __forceinline__ void wait_async0() {
  asm volatile("s_wait_asynccnt 0x0" ::: "memory");
}

// ---------------- init: zero output, reset routing state ----------------
__global__ __launch_bounds__(256) void kinit(float* __restrict__ out,
                                             int* __restrict__ stok,
                                             int* __restrict__ sexp,
                                             int* __restrict__ counts,
                                             int* __restrict__ fill) {
  int i = blockIdx.x * 256 + threadIdx.x;
  if (i < TOK * HID) out[i] = 0.0f;
  if (i < SLOT_MAX) { stok[i] = -1; sexp[i] = 0; }
  if (i < E_) { counts[i] = 0; fill[i] = 0; }
}

// ---------------- fp32 -> bf16 activation conversion ----------------
__global__ __launch_bounds__(256) void kconv(const float* __restrict__ x,
                                             __bf16* __restrict__ xb) {
  size_t i = ((size_t)blockIdx.x * 256 + threadIdx.x) * 8;
  float4 f0 = *(const float4*)(x + i);
  float4 f1 = *(const float4*)(x + i + 4);
  union { uint4 q; __bf16 h[8]; } o;
  o.h[0]=(__bf16)f0.x; o.h[1]=(__bf16)f0.y; o.h[2]=(__bf16)f0.z; o.h[3]=(__bf16)f0.w;
  o.h[4]=(__bf16)f1.x; o.h[5]=(__bf16)f1.y; o.h[6]=(__bf16)f1.z; o.h[7]=(__bf16)f1.w;
  *(uint4*)(xb + i) = o.q;
}

// ---------------- router: logits -> softmax -> top-2 -> renorm ----------------
__global__ __launch_bounds__(256) void krouter(const float* __restrict__ x,
                                               const float* __restrict__ gate,
                                               float* __restrict__ tkw,
                                               int* __restrict__ tki,
                                               int* __restrict__ counts) {
  __shared__ float glf[E_ * HID];                 // 64 KB of 320 KB WGP LDS
  const int tid = threadIdx.x;
  for (int i = tid * 4; i < E_ * HID; i += 1024)
    *(float4*)&glf[i] = *(const float4*)&gate[i];
  __syncthreads();

  const int lane = tid & 31, wid = tid >> 5;
  const int t = blockIdx.x * 8 + wid;             // one wave32 per token
  float acc[E_];
  #pragma unroll
  for (int e = 0; e < E_; ++e) acc[e] = 0.0f;
  for (int k = lane; k < HID; k += 32) {
    float xv = x[(size_t)t * HID + k];
    #pragma unroll
    for (int e = 0; e < E_; ++e) acc[e] += xv * glf[e * HID + k];
  }
  #pragma unroll
  for (int off = 16; off > 0; off >>= 1) {
    #pragma unroll
    for (int e = 0; e < E_; ++e) acc[e] += __shfl_down(acc[e], off);
  }
  if (lane == 0) {
    float mx = acc[0];
    #pragma unroll
    for (int e = 1; e < E_; ++e) mx = fmaxf(mx, acc[e]);
    float p[E_];
    #pragma unroll
    for (int e = 0; e < E_; ++e) p[e] = __expf(acc[e] - mx);
    int i0 = 0; float b0 = p[0];
    #pragma unroll
    for (int e = 1; e < E_; ++e) if (p[e] > b0) { b0 = p[e]; i0 = e; }
    int i1 = (i0 == 0) ? 1 : 0; float b1 = p[i1];
    #pragma unroll
    for (int e = 0; e < E_; ++e)
      if (e != i0 && p[e] > b1) { b1 = p[e]; i1 = e; }
    float rs = 1.0f / (b0 + b1);                  // softmax denom cancels
    tki[t * 2 + 0] = i0; tkw[t * 2 + 0] = b0 * rs;
    tki[t * 2 + 1] = i1; tkw[t * 2 + 1] = b1 * rs;
    atomicAdd(&counts[i0], 1);
    atomicAdd(&counts[i1], 1);
  }
}

// ---------------- scan: 64-aligned per-expert offsets (moe-align) ----------------
__global__ void kscan(const int* __restrict__ counts,
                      int* __restrict__ aoff, int* __restrict__ meta) {
  int off = 0;
  for (int e = 0; e < E_; ++e) {
    aoff[e] = off;
    off += ((counts[e] + MT - 1) / MT) * MT;
  }
  aoff[E_] = off;
  meta[0] = off;                                  // padded total
}

// ---------------- scatter tokens into expert slot lists ----------------
__global__ __launch_bounds__(256) void kscatter(const int* __restrict__ tki,
                                                const float* __restrict__ tkw,
                                                const int* __restrict__ aoff,
                                                int* __restrict__ fill,
                                                int* __restrict__ stok,
                                                int* __restrict__ sexp,
                                                float* __restrict__ sw) {
  int t = blockIdx.x * 256 + threadIdx.x;
  if (t >= TOK) return;
  #pragma unroll
  for (int k = 0; k < TOPK; ++k) {
    int e = tki[t * 2 + k];
    int pos = aoff[e] + atomicAdd(&fill[e], 1);
    stok[pos] = t; sexp[pos] = e; sw[pos] = tkw[t * 2 + k];
  }
}

// ---------------- K1: h = silu(x W1^T) * (x W3^T)  (gathered, per chunk) ------
__global__ __launch_bounds__(256) void kgate(const __bf16* __restrict__ xb,
                                             const float* __restrict__ w1,
                                             const float* __restrict__ w3,
                                             const int* __restrict__ stok,
                                             const int* __restrict__ sexp,
                                             const int* __restrict__ meta,
                                             __bf16* __restrict__ act,
                                             int s0base) {
  const int total = meta[0];
  const int s0 = s0base + blockIdx.x * MT;
  if (s0 >= total) return;
  const int e = sexp[s0];                          // tiles never cross experts
  const float* W1 = w1 + (size_t)e * ITR * HID;
  const float* W3 = w3 + (size_t)e * ITR * HID;
  const int n0 = blockIdx.y * NT;

  __shared__ __bf16 xt[MT][KB + 8];                // A slab, +8 halves pad
  __shared__ __bf16 bt1[NT][KB + 8];               // W1 slab (bf16)
  __shared__ __bf16 bt3[NT][KB + 8];               // W3 slab (bf16)
  __shared__ int ltok[MT];

  const int tid = threadIdx.x, lane = tid & 31, wid = tid >> 5;
  if (tid < MT) { int s = s0 + tid; ltok[tid] = (s < total) ? stok[s] : -1; }

  // A loader mapping: 64 rows x 4 eight-half segs (16B each)
  const int ldr = tid >> 2, seg = tid & 3;
  uint4 z = {0u, 0u, 0u, 0u};
  *(uint4*)&xt[ldr][seg * 8] = z;                  // invalid rows stay zero
  __syncthreads();

  const int t_row = ltok[ldr];
  const uint32_t alds = (uint32_t)(uintptr_t)&xt[ldr][seg * 8];

  // B stage mapping: 128 cols x 2 sixteen-float segs
  const int bcol = tid >> 1, bseg = (tid & 1) * 16;
  const float* wp1 = W1 + (size_t)(n0 + bcol) * HID + bseg;
  const float* wp3 = W3 + (size_t)(n0 + bcol) * HID + bseg;

  const int wm = (wid >> 2) * 32, wn = (wid & 3) * 32;
  const v8f vz = {0.f,0.f,0.f,0.f,0.f,0.f,0.f,0.f};
  v8f accg[2][2], accu[2][2];
  #pragma unroll
  for (int m = 0; m < 2; ++m)
    #pragma unroll
    for (int n = 0; n < 2; ++n) { accg[m][n] = vz; accu[m][n] = vz; }

  const int ar = lane & 15, koff = (lane >> 4) << 4;  // A frag: contiguous k = 16*(lane/16) + s
  const int bk = (lane >> 4) << 4;                    // B frag K-half

  for (int kk = 0; kk < HID; kk += KB) {
    // async gather of 64x32 bf16 A slab straight into LDS
    if (t_row >= 0)
      async_b128(alds, (uint64_t)(uintptr_t)(xb + (size_t)t_row * HID + kk + seg * 8));
    // cooperative fp32->bf16 conversion of both weight slabs into LDS
    { v16bf b = cvt16(wp1 + kk); BFQ u; u.v = b;
      *(uint4*)&bt1[bcol][bseg] = u.q[0]; *(uint4*)&bt1[bcol][bseg + 8] = u.q[1]; }
    { v16bf b = cvt16(wp3 + kk); BFQ u; u.v = b;
      *(uint4*)&bt3[bcol][bseg] = u.q[0]; *(uint4*)&bt3[bcol][bseg + 8] = u.q[1]; }
    if (kk + KB < HID) {
      __builtin_prefetch(wp1 + kk + KB, 0, 1);
      __builtin_prefetch(wp3 + kk + KB, 0, 1);
    }
    wait_async0();
    __syncthreads();

    v16bf a[2];
    #pragma unroll
    for (int m = 0; m < 2; ++m) {
      const __bf16* ap = &xt[wm + m * 16 + ar][koff];
      BFQ u; u.q[0] = *(const uint4*)ap; u.q[1] = *(const uint4*)(ap + 8);
      a[m] = u.v;
    }
    #pragma unroll
    for (int n = 0; n < 2; ++n) {
      const __bf16* b1p = &bt1[wn + n * 16 + (lane & 15)][bk];
      const __bf16* b3p = &bt3[wn + n * 16 + (lane & 15)][bk];
      BFQ u1; u1.q[0] = *(const uint4*)b1p; u1.q[1] = *(const uint4*)(b1p + 8);
      BFQ u3; u3.q[0] = *(const uint4*)b3p; u3.q[1] = *(const uint4*)(b3p + 8);
      #pragma unroll
      for (int m = 0; m < 2; ++m) {
        accg[m][n] = __builtin_amdgcn_wmma_f32_16x16x32_bf16(
            false, a[m], false, u1.v, (short)0, accg[m][n], false, false);
        accu[m][n] = __builtin_amdgcn_wmma_f32_16x16x32_bf16(
            false, a[m], false, u3.v, (short)0, accu[m][n], false, false);
      }
    }
    __syncthreads();
  }

  // epilogue: silu(g)*u -> bf16 act (C layout: VGPR i -> M = i + 8*(lane>=16))
  #pragma unroll
  for (int m = 0; m < 2; ++m)
    #pragma unroll
    for (int n = 0; n < 2; ++n)
      #pragma unroll
      for (int i = 0; i < 8; ++i) {
        float g = accg[m][n][i], uu = accu[m][n][i];
        float h = (g / (1.0f + __expf(-g))) * uu;
        int row = wm + m * 16 + ((lane >> 4) << 3) + i;
        int col = n0 + wn + n * 16 + (lane & 15);
        act[(size_t)(blockIdx.x * MT + row) * ITR + col] = (__bf16)h;
      }
}

// ---------------- K2: y = h W2^T, scatter-add w*y into out ----------------
__global__ __launch_bounds__(256) void kdown(const __bf16* __restrict__ act,
                                             const float* __restrict__ w2,
                                             const int* __restrict__ stok,
                                             const int* __restrict__ sexp,
                                             const float* __restrict__ sw,
                                             const int* __restrict__ meta,
                                             float* __restrict__ out,
                                             int s0base) {
  const int total = meta[0];
  const int s0 = s0base + blockIdx.x * MT;
  if (s0 >= total) return;
  const int e = sexp[s0];
  const float* W2 = w2 + (size_t)e * HID * ITR;
  const int n0 = blockIdx.y * NT;

  __shared__ __bf16 xt[MT][KB + 8];
  __shared__ __bf16 bt[NT][KB + 8];
  __shared__ int ltok[MT];
  __shared__ float lw[MT];
  const int tid = threadIdx.x, lane = tid & 31, wid = tid >> 5;
  if (tid < MT) {
    int s = s0 + tid;
    int t = (s < total) ? stok[s] : -1;
    ltok[tid] = t;
    lw[tid] = (s < total && t >= 0) ? sw[s] : 0.0f;
  }
  __syncthreads();

  const int wm = (wid >> 2) * 32, wn = (wid & 3) * 32;
  const int bcol = tid >> 1, bseg = (tid & 1) * 16;
  const float* wp = W2 + (size_t)(n0 + bcol) * ITR + bseg;

  const v8f vz = {0.f,0.f,0.f,0.f,0.f,0.f,0.f,0.f};
  v8f acc[2][2];
  #pragma unroll
  for (int m = 0; m < 2; ++m)
    #pragma unroll
    for (int n = 0; n < 2; ++n) acc[m][n] = vz;

  const int ar = lane & 15, koff = (lane >> 4) << 4;
  const int bk = (lane >> 4) << 4;
  const int ldr = tid >> 2, seg = tid & 3;
  const uint32_t xlds = (uint32_t)(uintptr_t)&xt[0][0];
  const uint64_t abase = (uint64_t)(uintptr_t)(act + (size_t)(blockIdx.x * MT) * ITR);

  for (int kk = 0; kk < ITR; kk += KB) {
#if __has_builtin(__builtin_amdgcn_tensor_load_to_lds)
    // Tensor Data Mover: 2D tile (32 bf16 x 64 rows) from act -> LDS,
    // pad 4 DWORDs after every 16 DWORDs to reproduce the [KB+8] LDS pitch.
    if (wid == 0) {
      u32x4 g0;
      uint64_t ga = abase + (uint64_t)kk * 2;
      g0[0] = 1u;                                  // count=1 valid descriptor
      g0[1] = xlds;                                // lds_addr
      g0[2] = (uint32_t)ga;                        // global_addr[31:0]
      g0[3] = (uint32_t)((ga >> 32) & 0x01ffffffu) | (2u << 30);  // addr[56:32] | type=2
      i32x8 g1;
      g1[0] = (1 << 16) | (1 << 20) | (3 << 22) | (3 << 25); // 2B elems, pad on, 16dw/4dw
      g1[1] = (int)((ITR & 0xffff) << 16);         // tensor_dim0 = ITR
      g1[2] = (int)((ITR >> 16) & 0xffff) | (int)((CHS & 0xffff) << 16); // dim0 hi | dim1
      g1[3] = (int)((CHS >> 16) & 0xffff) | (KB << 16);  // dim1 hi | tile_dim0 = 32
      g1[4] = MT;                                  // tile_dim1 = 64, tile_dim2 = 0
      g1[5] = ITR;                                 // tensor_dim0_stride (48b, lo)
      g1[6] = 0;                                   // stride hi | dim1_stride lo
      g1[7] = 0;
      i32x4 gz4 = {0, 0, 0, 0};
      i32x8 gz8 = {0, 0, 0, 0, 0, 0, 0, 0};
      __builtin_amdgcn_tensor_load_to_lds(g0, g1, gz4, gz4, gz8, 0);
      __builtin_amdgcn_s_wait_tensorcnt((short)0);
    }
#else
    async_b128((uint32_t)(uintptr_t)&xt[ldr][seg * 8],
               abase + ((uint64_t)ldr * ITR + kk + seg * 8) * 2);
#endif
    { v16bf b = cvt16(wp + kk); BFQ u; u.v = b;
      *(uint4*)&bt[bcol][bseg] = u.q[0]; *(uint4*)&bt[bcol][bseg + 8] = u.q[1]; }
    if (kk + KB < ITR) __builtin_prefetch(wp + kk + KB, 0, 1);
#if !__has_builtin(__builtin_amdgcn_tensor_load_to_lds)
    wait_async0();
#endif
    __syncthreads();

    v16bf a[2];
    #pragma unroll
    for (int m = 0; m < 2; ++m) {
      const __bf16* ap = &xt[wm + m * 16 + ar][koff];
      BFQ u; u.q[0] = *(const uint4*)ap; u.q[1] = *(const uint4*)(ap + 8);
      a[m] = u.v;
    }
    #pragma unroll
    for (int n = 0; n < 2; ++n) {
      const __bf16* bp = &bt[wn + n * 16 + (lane & 15)][bk];
      BFQ ub; ub.q[0] = *(const uint4*)bp; ub.q[1] = *(const uint4*)(bp + 8);
      #pragma unroll
      for (int m = 0; m < 2; ++m)
        acc[m][n] = __builtin_amdgcn_wmma_f32_16x16x32_bf16(
            false, a[m], false, ub.v, (short)0, acc[m][n], false, false);
    }
    __syncthreads();
  }

  // scatter-add weighted y (two contributions per output element overall)
  #pragma unroll
  for (int m = 0; m < 2; ++m)
    #pragma unroll
    for (int n = 0; n < 2; ++n)
      #pragma unroll
      for (int i = 0; i < 8; ++i) {
        int row = wm + m * 16 + ((lane >> 4) << 3) + i;
        int col = n0 + wn + n * 16 + (lane & 15);
        int t = ltok[row];
        if (t >= 0)
          unsafeAtomicAdd(&out[(size_t)t * HID + col], lw[row] * acc[m][n][i]);
      }
}

// ---------------- launch ----------------
extern "C" void kernel_launch(void* const* d_in, const int* in_sizes, int n_in,
                              void* d_out, int out_size, void* d_ws, size_t ws_size,
                              hipStream_t stream) {
  const float* x    = (const float*)d_in[0];   // [4,2048,2048]
  const float* gate = (const float*)d_in[1];   // [8,2048]
  const float* w1   = (const float*)d_in[2];   // [8,7168,2048]
  const float* w3   = (const float*)d_in[3];   // [8,7168,2048]
  const float* w2   = (const float*)d_in[4];   // [8,2048,7168]
  float* out = (float*)d_out;

  char* p = (char*)d_ws;
  auto take = [&](size_t b) -> char* {
    char* r = p; p += (b + 255) & ~(size_t)255; return r;
  };
  __bf16* xb   = (__bf16*)take((size_t)TOK * HID * 2);     // 33.6 MB
  float* tkw   = (float*) take((size_t)TOK * 2 * 4);
  int*   tki   = (int*)   take((size_t)TOK * 2 * 4);
  int*   counts= (int*)   take(E_ * 4);
  int*   aoff  = (int*)   take((E_ + 1) * 4);
  int*   fill  = (int*)   take(E_ * 4);
  int*   meta  = (int*)   take(4);
  int*   stok  = (int*)   take((size_t)SLOT_MAX * 4);
  int*   sexp  = (int*)   take((size_t)SLOT_MAX * 4);
  float* sw    = (float*) take((size_t)SLOT_MAX * 4);
  __bf16* act  = (__bf16*)take((size_t)CHS * ITR * 2);     // 29.4 MB
  if ((size_t)(p - (char*)d_ws) > ws_size) return;         // need ~64 MB scratch

  kinit<<<(TOK * HID + 255) / 256, 256, 0, stream>>>(out, stok, sexp, counts, fill);
  kconv<<<(TOK * HID / 8 + 255) / 256, 256, 0, stream>>>(x, xb);
  krouter<<<TOK / 8, 256, 0, stream>>>(x, gate, tkw, tki, counts);
  kscan<<<1, 1, 0, stream>>>(counts, aoff, meta);
  kscatter<<<(TOK + 255) / 256, 256, 0, stream>>>(tki, tkw, aoff, fill, stok, sexp, sw);

  for (int c = 0; c < NCHUNK; ++c) {
    int s0 = c * CHS;
    kgate<<<dim3(CHS / MT, ITR / NT), 256, 0, stream>>>(xb, w1, w3, stok, sexp, meta, act, s0);
    kdown<<<dim3(CHS / MT, HID / NT), 256, 0, stream>>>(act, w2, stok, sexp, sw, meta, out, s0);
  }
}
